// DynamicMaskHead_34557306864038
// MI455X (gfx1250) — compile-verified, matrix operand order checked by reading, not run on hardware
//
#include <hip/hip_runtime.h>

typedef __attribute__((ext_vector_type(16))) _Float16 v16h;
typedef __attribute__((ext_vector_type(8)))  float    v8f;

#define CH      8      // hidden channels
#define KIN     10     // 2 coords + 8 in channels
#define HH      128
#define WW      192
#define HWSZ    (HH * WW)
#define PARAMS  169
#define OFF_W0  0      // 8 x 10  (per row: [wx, wy, wf0..wf7])
#define OFF_W1  80     // 8 x 8
#define OFF_W2  144    // 1 x 8
#define OFF_B0  152
#define OFF_B1  160
#define OFF_B2  168

// One wave per (instance, row). Lanes 0-15 carry pixel N=lane (cols 16t..16t+15)
// and the K=0..7 half of operands; lanes 16-31 carry K=8..15.
// K permutation (consistent between A and B): K=0..7 = feature channels 0..7,
// K=8 = rel_x, K=9 = rel_y. This makes lanes 0-15 do a uniform 8-load feature
// fetch and lanes 16-31 compute coords arithmetically (no loads, no divergence
// in the B build).
__global__ __launch_bounds__(256) void mask_head_wmma(
    const float* __restrict__ mask_feats,     // (N, 8, H, W)
    const float* __restrict__ params,         // (n, 169)
    const float* __restrict__ inst_locs,      // (n, 2)
    const float* __restrict__ soi_tab,        // (6,)
    const int*   __restrict__ im_inds,        // (n,)
    const int*   __restrict__ fpn_levels,     // (n,)
    float*       __restrict__ logits,         // (n, H, W) scratch
    int n_inst)
{
  const int lane = threadIdx.x & 31;
  const int wave = threadIdx.x >> 5;
  const int pair = blockIdx.x * 8 + wave;         // uniform per wave
  if (pair >= n_inst * HH) return;
  const int inst = pair / HH;
  const int row  = pair % HH;

  const int half = lane >> 4;                     // 0: K/M 0..7, 1: K/M 8..15
  const int m    = lane & 15;                     // A row / pixel-in-tile

  const float* p = params + (size_t)inst * PARAMS;

  // ---- A matrices (weights) in 16x32 f16 layout, zero padded -------------
  v16h a0 = {};   // layer0 (K-permuted: feats at K=0..7, coords at K=8,9)
  v16h a1 = {};   // layer1: 8 x 8 weights (K=8..15 zero)
  if (m < CH) {
    const float* w0r = p + OFF_W0 + m * KIN;
    const float* w1r = p + OFF_W1 + m * CH;
    if (half == 0) {
      #pragma unroll
      for (int k = 0; k < 8; ++k) a0[k] = (_Float16)w0r[2 + k];  // feat weights
      #pragma unroll
      for (int k = 0; k < 8; ++k) a1[k] = (_Float16)w1r[k];
    } else {
      a0[0] = (_Float16)w0r[0];   // K=8: x-coord weight
      a0[1] = (_Float16)w0r[1];   // K=9: y-coord weight
    }
  }

  // ---- per-instance scalars; bias selects hoisted out of the loop --------
  float b0h[8], b1h[8], w2v[8];
  #pragma unroll
  for (int r = 0; r < 8; ++r) {
    b0h[r] = (half == 0) ? p[OFF_B0 + r] : 0.0f;   // C rows 8..15 stay 0
    b1h[r] = (half == 0) ? p[OFF_B1 + r] : 0.0f;
    w2v[r] = p[OFF_W2 + r];
  }
  const float b2v  = p[OFF_B2];
  const int   img  = im_inds[inst];
  const float invs = 1.0f / soi_tab[fpn_levels[inst]];
  const float ilx  = inst_locs[inst * 2 + 0];
  const float ily  = inst_locs[inst * 2 + 1];
  const float rely = (ily - (row * 8.0f + 4.0f)) * invs;

  const float* featbase = mask_feats + (size_t)img * (CH * HWSZ) + row * WW;
  float*       outrow   = logits + ((size_t)inst * HH + row) * WW;

  #pragma unroll 2
  for (int t = 0; t < WW / 16; ++t) {
    const int col = t * 16 + m;

    // ---- B0 build: half0 = 8 feature loads; half1 = computed coords ------
    v16h b0m = {};
    if (half == 0) {
      const float* fp = featbase + col;
      #pragma unroll
      for (int k = 0; k < 8; ++k) b0m[k] = (_Float16)fp[k * HWSZ];
    } else {
      float relx = (ilx - (col * 8.0f + 4.0f)) * invs;
      b0m[0] = (_Float16)relx;                    // K=8
      b0m[1] = (_Float16)rely;                    // K=9
    }

    // ---- layer 0 ----------------------------------------------------------
    v8f c0 = {};
    c0 = __builtin_amdgcn_wmma_f32_16x16x32_f16(false, a0, false, b0m,
                                                (short)0, c0, false, false);

    // bias + relu + in-lane repack into next-layer B:
    // half1 lanes hold C rows 8..15 (== 0) which map exactly to B's K=8..15.
    v16h h = {};
    #pragma unroll
    for (int r = 0; r < 8; ++r)
      h[r] = (_Float16)fmaxf(c0[r] + b0h[r], 0.0f);

    // ---- layer 1 ----------------------------------------------------------
    v8f c1 = {};
    c1 = __builtin_amdgcn_wmma_f32_16x16x32_f16(false, a1, false, h,
                                                (short)0, c1, false, false);

    // ---- layer 2: 1x8 dot + bias; lanes 0-15 hold channel r of pixel m ----
    float acc = b2v;
    #pragma unroll
    for (int r = 0; r < 8; ++r)
      acc = fmaf(w2v[r], fmaxf(c1[r] + b1h[r], 0.0f), acc);

    if (half == 0) outrow[col] = acc;
  }
}

// aligned_bilinear(factor=2): out(i,j) samples logits at ((max(i-1,0))/2, (max(j-1,0))/2)
__global__ __launch_bounds__(256) void upsample2x(
    const float* __restrict__ logits, float* __restrict__ out, int n_inst)
{
  const int OW = WW * 2, OH = HH * 2;
  long long idx = (long long)blockIdx.x * blockDim.x + threadIdx.x;
  long long total = (long long)n_inst * OH * OW;
  if (idx >= total) return;

  int j    = (int)(idx % OW);
  int tmp  = (int)(idx / OW);
  int i    = tmp % OH;
  int inst = tmp / OH;

  float y = (i == 0) ? 0.0f : (i - 1) * 0.5f;
  float x = (j == 0) ? 0.0f : (j - 1) * 0.5f;
  int   iy0 = (int)y;  float fy = y - (float)iy0;  int iy1 = min(iy0 + 1, HH - 1);
  int   ix0 = (int)x;  float fx = x - (float)ix0;  int ix1 = min(ix0 + 1, WW - 1);

  const float* src = logits + (size_t)inst * HWSZ;
  float v00 = src[iy0 * WW + ix0];
  float v01 = src[iy0 * WW + ix1];
  float v10 = src[iy1 * WW + ix0];
  float v11 = src[iy1 * WW + ix1];
  float v0 = v00 * (1.0f - fx) + v01 * fx;
  float v1 = v10 * (1.0f - fx) + v11 * fx;
  out[idx] = v0 * (1.0f - fy) + v1 * fy;
}

extern "C" void kernel_launch(void* const* d_in, const int* in_sizes, int n_in,
                              void* d_out, int out_size, void* d_ws, size_t ws_size,
                              hipStream_t stream) {
  const float* mask_feats = (const float*)d_in[0];
  const float* params     = (const float*)d_in[1];
  const float* inst_locs  = (const float*)d_in[2];
  const float* soi_tab    = (const float*)d_in[3];
  const int*   im_inds    = (const int*)d_in[4];
  const int*   fpn        = (const int*)d_in[5];

  const int n_inst = in_sizes[1] / PARAMS;   // 128
  float* logits = (float*)d_ws;              // n_inst * H * W floats (12.6 MB)

  const int pairs   = n_inst * HH;           // one wave per (inst, row)
  const int blocks1 = (pairs + 7) / 8;       // 8 waves per 256-thread block
  mask_head_wmma<<<blocks1, 256, 0, stream>>>(mask_feats, params, inst_locs,
                                              soi_tab, im_inds, fpn,
                                              logits, n_inst);

  const long long total = (long long)n_inst * (HH * 2) * (WW * 2);
  const int blocks2 = (int)((total + 255) / 256);
  upsample2x<<<blocks2, 256, 0, stream>>>(logits, (float*)d_out, n_inst);
}
